// CFConv_7928509628808
// MI455X (gfx1250) — matrix operand order repaired and checked by reading, not compiled, not run on hardware
//
#include <hip/hip_runtime.h>

// ---------------------------------------------------------------------------
// CFConv (SchNet) fused kernels for gfx1250, bf16 WMMA path.
//   roofline: ~340 MB mandatory HBM traffic (f_ij + indices) ~= 15us floor;
//   73 GFLOP of GEMM runs on v_wmma_f32_16x16x32_bf16; h/agg live in L2.
// ---------------------------------------------------------------------------

typedef __bf16 bf16;
typedef bf16  v8bf  __attribute__((ext_vector_type(8)));
typedef bf16  v16bf __attribute__((ext_vector_type(16)));
typedef float v8f   __attribute__((ext_vector_type(8)));

static constexpr int N_ATOMS   = 50000;
static constexpr int N_EDGES   = 1600000;
static constexpr int DIM       = 128;   // == N_FILTERS == OUT_DIM
static constexpr int N_GAUSS   = 50;
static constexpr int KGAUSS    = 64;    // padded gauss K for WMMA (mult of 32)
static constexpr int KS1       = KGAUSS + 8;  // 72 elems  (144 B row, 16B-mult)
static constexpr int KS2       = DIM + 8;     // 136 elems (272 B row, 16B-mult)
static constexpr float LOG2F_  = 0.69314718056f;
static constexpr float PI_OVER_CUT = 0.314159265358979f; // pi / 10
static constexpr float CUTOFF  = 10.0f;

__device__ __forceinline__ float sspf(float x) {
    float ax = __builtin_fabsf(x);
    float t  = __expf(-ax);
    return fmaxf(x, 0.0f) + __logf(1.0f + t) - LOG2F_;
}

__device__ __forceinline__ v8f wmma_bf16(v16bf a, v16bf b, v8f c) {
    return __builtin_amdgcn_wmma_f32_16x16x32_bf16(
        false, a, false, b, (short)0, c, false, false);
}

// A fragment 16x32 bf16 from LDS row-major tile [row][k].
__device__ __forceinline__ v16bf frag_a(const bf16* base, int stride, int kbase, int lane) {
    const int row = lane & 15;
    const int kb  = kbase + ((lane & 16) ? 8 : 0);
    const bf16* p = base + row * stride + kb;
    union { v16bf v; v8bf h[2]; } u;
    u.h[0] = *(const v8bf*)(p);
    u.h[1] = *(const v8bf*)(p + 16);
    return u.v;
}

// B fragment 32x16 bf16 from LDS transposed tile [n][k].
__device__ __forceinline__ v16bf frag_b(const bf16* baseT, int stride, int kbase, int lane) {
    const int col = lane & 15;
    const int kb  = kbase + ((lane & 16) ? 8 : 0);
    const bf16* p = baseT + col * stride + kb;
    union { v16bf v; v8bf h[2]; } u;
    u.h[0] = *(const v8bf*)(p);
    u.h[1] = *(const v8bf*)(p + 16);
    return u.v;
}

// ---------------------------------------------------------------------------
// Prep: bake Wf1/Wf2 into transposed, padded bf16 images (LDS-stride layout)
// so edge blocks stage weights with raw b128 copies.
// ---------------------------------------------------------------------------
__global__ __launch_bounds__(256) void k_prep(const float* __restrict__ Wf1,
                                              const float* __restrict__ Wf2,
                                              bf16* __restrict__ W1T,
                                              bf16* __restrict__ W2T) {
    const int stride = blockDim.x * gridDim.x;
    for (int idx = blockIdx.x * 256 + threadIdx.x; idx < 128 * KS1; idx += stride) {
        int n = idx / KS1, k = idx - n * KS1;
        W1T[idx] = (k < N_GAUSS) ? (bf16)Wf1[k * DIM + n] : (bf16)0.0f;
    }
    for (int idx = blockIdx.x * 256 + threadIdx.x; idx < 128 * KS2; idx += stride) {
        int n = idx / KS2, k = idx - n * KS2;
        W2T[idx] = (k < DIM) ? (bf16)Wf2[k * DIM + n] : (bf16)0.0f;
    }
}

// ---------------------------------------------------------------------------
// Kernel 1: h = x @ Win
// ---------------------------------------------------------------------------
__global__ __launch_bounds__(256) void k_project(const float* __restrict__ x,
                                                 const float* __restrict__ Win,
                                                 float* __restrict__ h,
                                                 int nrows) {
    __shared__ __align__(16) bf16 sX[128 * KS2];
    __shared__ __align__(16) bf16 sW[128 * KS2];   // Win^T : [n][k]
    const int r0  = blockIdx.x * 128;
    const int tid = threadIdx.x;
    const bool full = (r0 + 128 <= nrows);

    for (int idx = tid; idx < 128 * 128; idx += 256) {
        int a = idx >> 7, b = idx & 127;
        float xv = (full || r0 + a < nrows) ? x[(size_t)(r0 + a) * DIM + b] : 0.0f;
        sX[a * KS2 + b] = (bf16)xv;
        sW[b * KS2 + a] = (bf16)Win[a * DIM + b];
    }
    __syncthreads();

    const int wave = tid >> 5, lane = tid & 31;
    const bf16* Abase = sX + wave * 16 * KS2;
    v16bf a0 = frag_a(Abase, KS2, 0,  lane);
    v16bf a1 = frag_a(Abase, KS2, 32, lane);
    v16bf a2 = frag_a(Abase, KS2, 64, lane);
    v16bf a3 = frag_a(Abase, KS2, 96, lane);

    const int nlo   = lane & 15;
    const int rbase = wave * 16 + ((lane & 16) ? 8 : 0);
    for (int nt = 0; nt < 8; nt += 2) {
        const bf16* B0 = sW + (nt * 16) * KS2;
        const bf16* B1 = B0 + 16 * KS2;
        v8f acc0 = {}, acc1 = {};
        acc0 = wmma_bf16(a0, frag_b(B0, KS2, 0,  lane), acc0);
        acc1 = wmma_bf16(a0, frag_b(B1, KS2, 0,  lane), acc1);
        acc0 = wmma_bf16(a1, frag_b(B0, KS2, 32, lane), acc0);
        acc1 = wmma_bf16(a1, frag_b(B1, KS2, 32, lane), acc1);
        acc0 = wmma_bf16(a2, frag_b(B0, KS2, 64, lane), acc0);
        acc1 = wmma_bf16(a2, frag_b(B1, KS2, 64, lane), acc1);
        acc0 = wmma_bf16(a3, frag_b(B0, KS2, 96, lane), acc0);
        acc1 = wmma_bf16(a3, frag_b(B1, KS2, 96, lane), acc1);

        const int n0 = nt * 16 + nlo, n1 = n0 + 16;
        if (full) {
            #pragma unroll
            for (int r = 0; r < 8; ++r) {
                size_t row = (size_t)(r0 + rbase + r) * DIM;
                h[row + n0] = acc0[r];
                h[row + n1] = acc1[r];
            }
        } else {
            #pragma unroll
            for (int r = 0; r < 8; ++r) {
                int row = r0 + rbase + r;
                if (row < nrows) {
                    h[(size_t)row * DIM + n0] = acc0[r];
                    h[(size_t)row * DIM + n1] = acc1[r];
                }
            }
        }
    }
}

// ---------------------------------------------------------------------------
// Kernel 2 (fused edge pipeline)
// ---------------------------------------------------------------------------
__global__ __launch_bounds__(256) void k_edges(const float* __restrict__ f_ij,
                                               const float* __restrict__ r_ij,
                                               const int*   __restrict__ ind_i,
                                               const int*   __restrict__ ind_j,
                                               const bf16*  __restrict__ W1T,
                                               const float* __restrict__ bf1,
                                               const bf16*  __restrict__ W2T,
                                               const float* __restrict__ bf2,
                                               const float* __restrict__ h,
                                               float* __restrict__ agg) {
    __shared__ __align__(16) bf16 sF [128 * KS1];  // f_ij tile, K padded to 64
    __shared__ __align__(16) bf16 sW1[128 * KS1];  // Wf1^T [n][k]
    __shared__ __align__(16) bf16 sW2[128 * KS2];  // Wf2^T [n][k]
    __shared__ __align__(16) bf16 sT [128 * KS2];  // T1 [edge][channel]
    __shared__ float sC[128];
    __shared__ float sB1[128];
    __shared__ float sB2[128];
    __shared__ int   sI[128];
    __shared__ int   sJ[128];

    const int e0  = blockIdx.x * 128;
    const int tid = threadIdx.x;

    // --- stage: weights as raw b128 copies (pre-converted, pre-padded) ---
    {
        const v8bf* g1 = (const v8bf*)W1T;
        v8bf* l1 = (v8bf*)sW1;
        for (int idx = tid; idx < 128 * KS1 / 8; idx += 256) l1[idx] = g1[idx];
        const v8bf* g2 = (const v8bf*)W2T;
        v8bf* l2 = (v8bf*)sW2;
        for (int idx = tid; idx < 128 * KS2 / 8; idx += 256) l2[idx] = g2[idx];
    }
    // --- stage: f_ij tile, fully coalesced; pad columns zeroed separately ---
    for (int idx = tid; idx < 128 * N_GAUSS; idx += 256) {
        int r = idx / N_GAUSS, k = idx - r * N_GAUSS;
        sF[r * KS1 + k] = (bf16)f_ij[(size_t)e0 * N_GAUSS + idx];
    }
    for (int idx = tid; idx < 128 * 16; idx += 256) {
        int r = idx >> 4, k = 48 + (idx & 15);
        if (k >= N_GAUSS) sF[r * KS1 + k] = (bf16)0.0f;   // zero K 50..63
    }
    if (tid < 128) {
        int e   = e0 + tid;
        float r = r_ij[e];
        float c = 0.5f * (__cosf(r * PI_OVER_CUT) + 1.0f);
        sC[tid]  = (r < CUTOFF) ? c : 0.0f;
        sI[tid]  = ind_i[e];
        sJ[tid]  = ind_j[e];
        sB1[tid] = bf1[tid];
        sB2[tid] = bf2[tid];
    }
    // prefetch next block's f_ij tile while this block computes
    if (e0 + 128 < N_EDGES) {
        const char* nxt = (const char*)(f_ij + (size_t)(e0 + 128) * N_GAUSS);
        __builtin_prefetch(nxt + tid * 100, 0, 1);
    }
    __syncthreads();

    const int wave = tid >> 5, lane = tid & 31;
    const int rb   = wave * 16 + ((lane & 16) ? 8 : 0);
    const int nlo  = lane & 15;

    // --- GEMM1: [16 x 64] @ [64 x 128] -> ssp -> sT (bf16) ---
    const bf16* A1 = sF + wave * 16 * KS1;
    {
        v16bf a0 = frag_a(A1, KS1, 0,  lane);
        v16bf a1 = frag_a(A1, KS1, 32, lane);
        for (int nt = 0; nt < 8; nt += 2) {
            const bf16* B0 = sW1 + (nt * 16) * KS1;
            const bf16* B1 = B0 + 16 * KS1;
            v8f acc0 = {}, acc1 = {};
            acc0 = wmma_bf16(a0, frag_b(B0, KS1, 0,  lane), acc0);
            acc1 = wmma_bf16(a0, frag_b(B1, KS1, 0,  lane), acc1);
            acc0 = wmma_bf16(a1, frag_b(B0, KS1, 32, lane), acc0);
            acc1 = wmma_bf16(a1, frag_b(B1, KS1, 32, lane), acc1);
            const int n0 = nt * 16 + nlo, n1 = n0 + 16;
            const float b0v = sB1[n0], b1v = sB1[n1];
            #pragma unroll
            for (int r = 0; r < 8; ++r) {
                sT[(rb + r) * KS2 + n0] = (bf16)sspf(acc0[r] + b0v);
                sT[(rb + r) * KS2 + n1] = (bf16)sspf(acc1[r] + b1v);
            }
        }
    }
    // Each wave only re-reads its own 16 sT rows (per-wave DS ops in-order).

    // --- GEMM2 + ssp*cutoff + gather h + scatter-add agg ---
    const bf16* A2 = sT + wave * 16 * KS2;
    {
        v16bf a0 = frag_a(A2, KS2, 0,  lane);
        v16bf a1 = frag_a(A2, KS2, 32, lane);
        v16bf a2 = frag_a(A2, KS2, 64, lane);
        v16bf a3 = frag_a(A2, KS2, 96, lane);
        for (int nt = 0; nt < 8; nt += 2) {
            const bf16* B0 = sW2 + (nt * 16) * KS2;
            const bf16* B1 = B0 + 16 * KS2;
            v8f acc0 = {}, acc1 = {};
            acc0 = wmma_bf16(a0, frag_b(B0, KS2, 0,  lane), acc0);
            acc1 = wmma_bf16(a0, frag_b(B1, KS2, 0,  lane), acc1);
            acc0 = wmma_bf16(a1, frag_b(B0, KS2, 32, lane), acc0);
            acc1 = wmma_bf16(a1, frag_b(B1, KS2, 32, lane), acc1);
            acc0 = wmma_bf16(a2, frag_b(B0, KS2, 64, lane), acc0);
            acc1 = wmma_bf16(a2, frag_b(B1, KS2, 64, lane), acc1);
            acc0 = wmma_bf16(a3, frag_b(B0, KS2, 96, lane), acc0);
            acc1 = wmma_bf16(a3, frag_b(B1, KS2, 96, lane), acc1);

            const int n0 = nt * 16 + nlo, n1 = n0 + 16;
            const float b0v = sB2[n0], b1v = sB2[n1];
            #pragma unroll
            for (int r = 0; r < 8; ++r) {
                const int R = rb + r;
                const float c = sC[R];
                const float* hr = h + (size_t)sJ[R] * DIM;
                float* ar = agg + (size_t)sI[R] * DIM;
                float w0 = sspf(acc0[r] + b0v) * c;
                float w1 = sspf(acc1[r] + b1v) * c;
                __hip_atomic_fetch_add(ar + n0, w0 * hr[n0],
                                       __ATOMIC_RELAXED, __HIP_MEMORY_SCOPE_AGENT);
                __hip_atomic_fetch_add(ar + n1, w1 * hr[n1],
                                       __ATOMIC_RELAXED, __HIP_MEMORY_SCOPE_AGENT);
            }
        }
    }
}

// ---------------------------------------------------------------------------
// Kernel 3: out = ssp(agg @ Wout + bout)
// ---------------------------------------------------------------------------
__global__ __launch_bounds__(256) void k_output(const float* __restrict__ agg,
                                                const float* __restrict__ Wout,
                                                const float* __restrict__ bout,
                                                float* __restrict__ out,
                                                int nrows) {
    __shared__ __align__(16) bf16 sX[128 * KS2];
    __shared__ __align__(16) bf16 sW[128 * KS2];   // Wout^T
    const int r0  = blockIdx.x * 128;
    const int tid = threadIdx.x;
    const bool full = (r0 + 128 <= nrows);

    for (int idx = tid; idx < 128 * 128; idx += 256) {
        int a = idx >> 7, b = idx & 127;
        float xv = (full || r0 + a < nrows) ? agg[(size_t)(r0 + a) * DIM + b] : 0.0f;
        sX[a * KS2 + b] = (bf16)xv;
        sW[b * KS2 + a] = (bf16)Wout[a * DIM + b];
    }
    __syncthreads();

    const int wave = tid >> 5, lane = tid & 31;
    const bf16* Abase = sX + wave * 16 * KS2;
    v16bf a0 = frag_a(Abase, KS2, 0,  lane);
    v16bf a1 = frag_a(Abase, KS2, 32, lane);
    v16bf a2 = frag_a(Abase, KS2, 64, lane);
    v16bf a3 = frag_a(Abase, KS2, 96, lane);

    const int nlo   = lane & 15;
    const int rbase = wave * 16 + ((lane & 16) ? 8 : 0);
    for (int nt = 0; nt < 8; nt += 2) {
        const bf16* B0 = sW + (nt * 16) * KS2;
        const bf16* B1 = B0 + 16 * KS2;
        v8f acc0 = {}, acc1 = {};
        acc0 = wmma_bf16(a0, frag_b(B0, KS2, 0,  lane), acc0);
        acc1 = wmma_bf16(a0, frag_b(B1, KS2, 0,  lane), acc1);
        acc0 = wmma_bf16(a1, frag_b(B0, KS2, 32, lane), acc0);
        acc1 = wmma_bf16(a1, frag_b(B1, KS2, 32, lane), acc1);
        acc0 = wmma_bf16(a2, frag_b(B0, KS2, 64, lane), acc0);
        acc1 = wmma_bf16(a2, frag_b(B1, KS2, 64, lane), acc1);
        acc0 = wmma_bf16(a3, frag_b(B0, KS2, 96, lane), acc0);
        acc1 = wmma_bf16(a3, frag_b(B1, KS2, 96, lane), acc1);

        const int n0 = nt * 16 + nlo, n1 = n0 + 16;
        const float b0v = bout[n0], b1v = bout[n1];
        if (full) {
            #pragma unroll
            for (int r = 0; r < 8; ++r) {
                size_t row = (size_t)(r0 + rbase + r) * DIM;
                out[row + n0] = sspf(acc0[r] + b0v);
                out[row + n1] = sspf(acc1[r] + b1v);
            }
        } else {
            #pragma unroll
            for (int r = 0; r < 8; ++r) {
                int row = r0 + rbase + r;
                if (row < nrows) {
                    out[(size_t)row * DIM + n0] = sspf(acc0[r] + b0v);
                    out[(size_t)row * DIM + n1] = sspf(acc1[r] + b1v);
                }
            }
        }
    }
}

// ---------------------------------------------------------------------------
extern "C" void kernel_launch(void* const* d_in, const int* in_sizes, int n_in,
                              void* d_out, int out_size, void* d_ws, size_t ws_size,
                              hipStream_t stream) {
    const float* x     = (const float*)d_in[0];
    const float* r_ij  = (const float*)d_in[1];
    const float* f_ij  = (const float*)d_in[2];
    const int*   ind_i = (const int*)  d_in[3];
    const int*   ind_j = (const int*)  d_in[4];
    const float* Wf1   = (const float*)d_in[5];
    const float* bf1   = (const float*)d_in[6];
    const float* Wf2   = (const float*)d_in[7];
    const float* bf2   = (const float*)d_in[8];
    const float* Win   = (const float*)d_in[9];
    const float* Wout  = (const float*)d_in[10];
    const float* bout  = (const float*)d_in[11];

    float* out = (float*)d_out;
    float* agg = (float*)d_ws;                              // [N_ATOMS*DIM] f32
    float* h   = agg + (size_t)N_ATOMS * DIM;               // [N_ATOMS*DIM] f32
    bf16*  W1T = (bf16*)(h + (size_t)N_ATOMS * DIM);        // [128*KS1] bf16
    bf16*  W2T = W1T + 128 * KS1;                           // [128*KS2] bf16

    hipMemsetAsync(agg, 0, (size_t)N_ATOMS * DIM * sizeof(float), stream);

    const int nodeBlocks = (N_ATOMS + 127) / 128;           // 391
    const int edgeBlocks = N_EDGES / 128;                   // 12500 (exact)

    k_prep   <<<32, 256, 0, stream>>>(Wf1, Wf2, W1T, W2T);
    k_project<<<nodeBlocks, 256, 0, stream>>>(x, Win, h, N_ATOMS);
    k_edges  <<<edgeBlocks, 256, 0, stream>>>(f_ij, r_ij, ind_i, ind_j,
                                              W1T, bf1, W2T, bf2, h, agg);
    k_output <<<nodeBlocks, 256, 0, stream>>>(agg, Wout, bout, out, N_ATOMS);
}
